// DualEncoder_49907519980134
// MI455X (gfx1250) — compile-verified, matrix operand order checked by reading, not run on hardware
//
#include <hip/hip_runtime.h>
#include <hip/hip_bf16.h>
#include <math.h>

typedef __attribute__((ext_vector_type(16))) _Float16 v16h;
typedef __attribute__((ext_vector_type(8)))  _Float16 v8h;
typedef __attribute__((ext_vector_type(8)))  float    v8f;

#define FLAG_RELU   1
#define FLAG_F32OUT 2

union Frag16 { v16h v; v8h h[2]; };

// ---------------------------------------------------------------------------
// Templated WMMA GEMM:  out = epilogue(A[R,Kpad] @ B[Kpad,Nout])
//   A   : f16 row-major, stride Kpad (Kpad % 32 == 0, 16B aligned)
//   BT  : f16 [Nout, Kpad] (transposed weights, zero padded K)
//   scale/bias : f32 per-output-channel affine (scale may be null -> 1.0)
//   RT  : row tiles per block (rows = RT*16), compile-time (2,4,8)
//   MP  : fused per-block max-pool over all rows -> f32 [R/rows, Nout]
// grid = (R/(RT*16), Nout/64), block = 256 threads (8 waves)
// Software-pipelined by 2 with NAMED A fragments + fixed LDS buffers so all
// fragments stay in registers (no dynamic indexing -> no alloca demotion).
// ---------------------------------------------------------------------------
template <int RT, bool MP, bool RELU, bool OUTF32>
__global__ __launch_bounds__(256) void wmma_gemm_t(
    const _Float16* __restrict__ A, const _Float16* __restrict__ BT,
    const float* __restrict__ scale, const float* __restrict__ bias,
    void* __restrict__ outp, int Kpad, int Nout)
{
  constexpr int CW   = RT / 2;     // col tiles per wave
  constexpr int ROWS = RT * 16;

  __shared__ _Float16 ldsB0[64 * 40];  // 64 cols x 32 k-halves, stride 40
  __shared__ _Float16 ldsB1[64 * 40];
  __shared__ float    maxpart[8 * 64];

  const int tid  = threadIdx.x;
  const int w    = tid >> 5;
  const int lane = tid & 31;
  const int rt     = w % RT;
  const int ctbase = (w / RT) * CW;
  const int rowbase = blockIdx.x * ROWS;
  const int colbase = blockIdx.y * 64;
  const int m    = lane & 15;
  const int half = lane >> 4;

  v8f acc[CW];
#pragma unroll
  for (int c = 0; c < CW; ++c)
#pragma unroll
    for (int r = 0; r < 8; ++r) acc[c][r] = 0.0f;

  const int ldn  = tid >> 2;        // B stage: col per thread
  const int ldch = (tid & 3) * 8;   // B stage: k-offset per thread
  const int stoff = ldn * 40 + ldch;
  const int bfoff = (ctbase * 16 + m) * 40 + 8 * half;  // B frag base (per wave)
  const _Float16* Bcol = BT + (size_t)(colbase + ldn) * Kpad + ldch;
  const _Float16* Arow = A + (size_t)(rowbase + rt * 16 + m) * Kpad + 8 * half;

  // compute CW wmmas from one A fragment + one LDS B buffer
  auto compute = [&](const v16h& av, const _Float16* lb) {
#pragma unroll
    for (int c = 0; c < CW; ++c) {
      const _Float16* bp = lb + bfoff + c * 16 * 40;
      Frag16 bf;
      bf.h[0] = *(const v8h*)bp;
      bf.h[1] = *(const v8h*)(bp + 16);
      acc[c] = __builtin_amdgcn_wmma_f32_16x16x32_f16(
          false, av, false, bf.v, (short)0, acc[c], false, false);
    }
  };

  // prologue: A fragment 0 + stage B chunk 0 (ISA 16-bit A 16x32 layout:
  // each lane holds two contiguous 16B k-runs)
  Frag16 a0, a1;
  a0.h[0] = *(const v8h*)(Arow);
  a0.h[1] = *(const v8h*)(Arow + 16);
  *(v8h*)(&ldsB0[stoff]) = *(const v8h*)(Bcol);
  __syncthreads();

  const int nchunks = Kpad >> 5;
  int ck = 0;
  for (;;) {
    // even phase: consume a0 / ldsB0, prefetch chunk ck+1 into a1 / ldsB1
    if (ck + 1 < nchunks) {
      const int kb = (ck + 1) << 5;
      a1.h[0] = *(const v8h*)(Arow + kb);
      a1.h[1] = *(const v8h*)(Arow + kb + 16);
      *(v8h*)(&ldsB1[stoff]) = *(const v8h*)(Bcol + kb);
    }
    compute(a0.v, ldsB0);
    __syncthreads();
    if (ck + 1 >= nchunks) break;

    // odd phase: consume a1 / ldsB1, prefetch chunk ck+2 into a0 / ldsB0
    if (ck + 2 < nchunks) {
      const int kb = (ck + 2) << 5;
      a0.h[0] = *(const v8h*)(Arow + kb);
      a0.h[1] = *(const v8h*)(Arow + kb + 16);
      *(v8h*)(&ldsB0[stoff]) = *(const v8h*)(Bcol + kb);
    }
    compute(a1.v, ldsB1);
    __syncthreads();
    if (ck + 2 >= nchunks) break;
    ck += 2;
  }

#pragma unroll
  for (int c = 0; c < CW; ++c) {
    const int ct  = ctbase + c;
    const int col = colbase + ct * 16 + m;
    float sc = scale ? scale[col] : 1.0f;
    float bs = bias ? bias[col] : 0.0f;
    if (!MP) {
      float*    of = (float*)outp;
      _Float16* oh = (_Float16*)outp;
#pragma unroll
      for (int r = 0; r < 8; ++r) {
        float y = acc[c][r] * sc + bs;
        if (RELU) y = y > 0.0f ? y : 0.0f;
        size_t o = (size_t)(rowbase + rt * 16 + 8 * half + r) * Nout + col;
        if (OUTF32) of[o] = y; else oh[o] = (_Float16)y;
      }
    } else {
      float mx = -3.4e38f;
#pragma unroll
      for (int r = 0; r < 8; ++r) {
        float y = acc[c][r] * sc + bs;
        if (RELU) y = y > 0.0f ? y : 0.0f;
        mx = y > mx ? y : mx;
      }
      float o = __shfl_xor(mx, 16);
      mx = o > mx ? o : mx;
      if (half == 0) maxpart[rt * 64 + ct * 16 + m] = mx;
    }
  }
  if (MP) {
    __syncthreads();
    if (tid < 64) {
      float mx = maxpart[tid];
#pragma unroll
      for (int r2 = 1; r2 < RT; ++r2) {
        float v = maxpart[r2 * 64 + tid];
        mx = v > mx ? v : mx;
      }
      ((float*)outp)[(size_t)blockIdx.x * Nout + colbase + tid] = mx;
    }
  }
}

// ---------------------------------------------------------------------------
// Farthest point sampling, start index 0; one block per cloud
// ---------------------------------------------------------------------------
__global__ __launch_bounds__(256) void fps_kernel(const float* __restrict__ xyz,
                                                  int N, int S, int* __restrict__ outIdx)
{
  int g = blockIdx.x;
  const float* p = xyz + (size_t)g * N * 3;
  int t = threadIdx.x;
  float dloc[8];
#pragma unroll
  for (int k = 0; k < 8; ++k) dloc[k] = 1e10f;
  int cnt = (N + 255) >> 8;
  __shared__ float redV[8];
  __shared__ int   redI[8];
  __shared__ int   curS;
  int cur = 0;
  for (int it = 0; it < S; ++it) {
    if (t == 0) outIdx[(size_t)g * S + it] = cur;
    float cx = p[cur * 3 + 0], cy = p[cur * 3 + 1], cz = p[cur * 3 + 2];
    float bv = -1.0f; int bi = 0;
    for (int k = 0; k < cnt; ++k) {
      int i = t + (k << 8);
      if (i < N) {
        float dx = p[i * 3 + 0] - cx, dy = p[i * 3 + 1] - cy, dz = p[i * 3 + 2] - cz;
        float d = dx * dx + dy * dy + dz * dz;
        float dl = dloc[k];
        d = d < dl ? d : dl;
        dloc[k] = d;
        if (d > bv) { bv = d; bi = i; }
      }
    }
    for (int off = 16; off; off >>= 1) {
      float ov = __shfl_down(bv, off);
      int   oi = __shfl_down(bi, off);
      if (ov > bv || (ov == bv && oi < bi)) { bv = ov; bi = oi; }
    }
    if ((t & 31) == 0) { redV[t >> 5] = bv; redI[t >> 5] = bi; }
    __syncthreads();
    if (t == 0) {
      float best = redV[0]; int besti = redI[0];
      for (int q = 1; q < 8; ++q) {
        if (redV[q] > best || (redV[q] == best && redI[q] < besti)) { best = redV[q]; besti = redI[q]; }
      }
      curS = besti;
    }
    __syncthreads();
    cur = curS;
  }
}

__global__ void gather_xyz_kernel(const float* __restrict__ xyz, const int* __restrict__ fpsIdx,
                                  int N, int S, float* __restrict__ newxyz, int GC)
{
  int i = blockIdx.x * blockDim.x + threadIdx.x;
  if (i >= GC) return;
  int g = i / S;
  int id = fpsIdx[i];
  const float* p = xyz + ((size_t)g * N + id) * 3;
  float* o = newxyz + (size_t)i * 3;
  o[0] = p[0]; o[1] = p[1]; o[2] = p[2];
}

// ---------------------------------------------------------------------------
// Ball query: one wave per center; first-K in-radius neighbors in index order
// ---------------------------------------------------------------------------
__global__ __launch_bounds__(256) void ballquery_kernel(
    const float* __restrict__ xyz, const float* __restrict__ newxyz,
    int N, int S, int K, float r2, int* __restrict__ outIdx, int GC)
{
  int wid  = (blockIdx.x * blockDim.x + threadIdx.x) >> 5;
  int lane = threadIdx.x & 31;
  if (wid >= GC) return;
  int g = wid / S;
  const float* p = xyz + (size_t)g * N * 3;
  float cx = newxyz[(size_t)wid * 3 + 0];
  float cy = newxyz[(size_t)wid * 3 + 1];
  float cz = newxyz[(size_t)wid * 3 + 2];
  int* out = outIdx + (size_t)wid * K;
  int base = 0;
  for (int i0 = 0; i0 < N && base < K; i0 += 32) {
    int i = i0 + lane;
    bool pred = false;
    if (i < N) {
      float dx = p[i * 3 + 0] - cx, dy = p[i * 3 + 1] - cy, dz = p[i * 3 + 2] - cz;
      pred = (dx * dx + dy * dy + dz * dz) < r2;
    }
    unsigned mask = (unsigned)__ballot(pred);
    int pre = __popc(mask & ((1u << lane) - 1u));
    if (pred && (base + pre) < K) out[base + pre] = i;
    base += __popc(mask);
  }
  int filled = base < K ? base : K;
  for (int j = filled + lane; j < K; j += 32) out[j] = 0;
}

// ---------------------------------------------------------------------------
// Grouping: build grouped f16 input rows [GC*K, CinPad]
//   row = [xyz[idx]-center (3), feats[idx] (Cf), zeros pad]
// ---------------------------------------------------------------------------
__global__ __launch_bounds__(64) void group_kernel(
    const float* __restrict__ xyz, const float* __restrict__ feats,
    const float* __restrict__ newxyz, const int* __restrict__ idx,
    int N, int S, int K, int Cf, int CinPad, _Float16* __restrict__ out)
{
  int row = blockIdx.x;                 // GC*K
  int k = row % K;
  int cs = row / K;
  int g = cs / S;
  int id = idx[(size_t)cs * K + k];
  const float* pp = xyz + ((size_t)g * N + id) * 3;
  const float* c  = newxyz + (size_t)cs * 3;
  _Float16* o = out + (size_t)row * CinPad;
  int t = threadIdx.x;
  if (t < 3) o[t] = (_Float16)(pp[t] - c[t]);
  for (int j = t; j < CinPad - 3; j += 64) {
    float v = (j < Cf) ? feats[((size_t)g * N + id) * Cf + j] : 0.0f;
    o[3 + j] = (_Float16)v;
  }
}

// ---------------------------------------------------------------------------
// Weight prep: WT[n, k] = (k < Cin ? W[k, n] : 0) in f16
// ---------------------------------------------------------------------------
__global__ void prep_wt_kernel(const float* __restrict__ W, int Cin, int Cout, int Kpad,
                               _Float16* __restrict__ WT)
{
  int i = blockIdx.x * blockDim.x + threadIdx.x;
  if (i >= Cout * Kpad) return;
  int n = i / Kpad, k = i - n * Kpad;
  float v = (k < Cin) ? W[(size_t)k * Cout + n] : 0.0f;
  WT[i] = (_Float16)v;
}

__global__ void prep_bn_kernel(const float* __restrict__ b, const float* __restrict__ g,
                               const float* __restrict__ beta, const float* __restrict__ mean,
                               const float* __restrict__ var, int C,
                               float* __restrict__ scale, float* __restrict__ bias)
{
  int i = blockIdx.x * blockDim.x + threadIdx.x;
  if (i >= C) return;
  float s = g[i] * rsqrtf(var[i] + 1e-5f);
  scale[i] = s;
  bias[i]  = (b[i] - mean[i]) * s + beta[i];
}

// ---------------------------------------------------------------------------
// LayerNorm + ReLU, f32 in -> f16 out, one block per row
// ---------------------------------------------------------------------------
__global__ __launch_bounds__(256) void ln_relu_kernel(const float* __restrict__ X,
                                                      const float* __restrict__ g,
                                                      const float* __restrict__ b,
                                                      int C, _Float16* __restrict__ out)
{
  int row = blockIdx.x;
  const float* x = X + (size_t)row * C;
  __shared__ float rs[8], rs2[8];
  __shared__ float smu, sinv;
  float s = 0.0f, s2 = 0.0f;
  for (int i = threadIdx.x; i < C; i += 256) { float v = x[i]; s += v; s2 += v * v; }
  for (int off = 16; off; off >>= 1) { s += __shfl_down(s, off); s2 += __shfl_down(s2, off); }
  if ((threadIdx.x & 31) == 0) { rs[threadIdx.x >> 5] = s; rs2[threadIdx.x >> 5] = s2; }
  __syncthreads();
  if (threadIdx.x == 0) {
    float S = 0.0f, S2 = 0.0f;
    for (int q = 0; q < 8; ++q) { S += rs[q]; S2 += rs2[q]; }
    float mu = S / C;
    smu = mu;
    sinv = rsqrtf(S2 / C - mu * mu + 1e-5f);
  }
  __syncthreads();
  float mu = smu, inv = sinv;
  _Float16* o = out + (size_t)row * C;
  for (int i = threadIdx.x; i < C; i += 256) {
    float y = (x[i] - mu) * inv * g[i] + b[i];
    o[i] = (_Float16)(y > 0.0f ? y : 0.0f);
  }
}

// ---------------------------------------------------------------------------
// Fourier PE + mask_cond assembly: cond[r] = [mask_token(1024), pe(24), pad]
// ---------------------------------------------------------------------------
#define CONDPAD 1056
__global__ __launch_bounds__(32) void pe_kernel(const float* __restrict__ centers,
                                                const float* __restrict__ peW,
                                                const float* __restrict__ peB,
                                                const float* __restrict__ mtok,
                                                _Float16* __restrict__ cond, int BM)
{
  int r = blockIdx.x;
  if (r >= BM) return;
  __shared__ float f[24];
  int t = threadIdx.x;
  if (t < 24) {
    int d = t / 8, j = t % 8, band = j & 3;
    float x = centers[r * 3 + d];
    float ang = x * exp2f((float)band) * 3.14159265358979323846f;
    f[t] = (j >= 4) ? cosf(ang) : sinf(ang);
  }
  __syncthreads();
  _Float16* o = cond + (size_t)r * CONDPAD;
  for (int i = t; i < 1024; i += 32) o[i] = (_Float16)mtok[i];
  if (t < 24) {
    float acc = peB[t];
    for (int k = 0; k < 24; ++k) acc += f[k] * peW[k * 24 + t];
    o[1024 + t] = (_Float16)acc;
  }
  for (int i = 1048 + t; i < CONDPAD; i += 32) o[i] = (_Float16)0.0f;
}

__global__ void cvt_f16_kernel(const float* __restrict__ a, _Float16* __restrict__ b, int n)
{
  int i = blockIdx.x * blockDim.x + threadIdx.x;
  if (i < n) b[i] = (_Float16)a[i];
}

__global__ void copy_f32_kernel(const float* __restrict__ a, float* __restrict__ b, int n)
{
  int i = blockIdx.x * blockDim.x + threadIdx.x;
  if (i < n) b[i] = a[i];
}

// pred[b,m,p,c] = h[b,p,c] * sigmoid(gpre[b,m,c])
__global__ __launch_bounds__(256) void combine_kernel(const float* __restrict__ h,
                                                      const float* __restrict__ gpre,
                                                      float* __restrict__ out)
{
  int i = blockIdx.x * 256 + threadIdx.x;   // 8*4*32*1024 total
  int c = i & 1023;
  int p = (i >> 10) & 31;
  int m = (i >> 15) & 3;
  int b = i >> 17;
  float gv = gpre[(size_t)((b << 2) + m) * 1024 + c];
  float sig = 1.0f / (1.0f + expf(-gv));
  out[i] = h[(size_t)((b << 5) + p) * 1024 + c] * sig;
}

// ---------------------------------------------------------------------------
// Host-side orchestration
// ---------------------------------------------------------------------------
struct SAConf { int S, K, Cin, CinPad, ch0, ch1, ch2; float r2; };
static const SAConf g_sa[3] = {
  {512,  32,   3,  32,   64,  64,  128, 0.01f},
  {128,  64, 131, 160,  128, 128,  256, 0.0625f},
  { 32, 128, 259, 288,  256, 512, 1024, 0.25f},
};

struct LayerW {
  const float *W, *b, *g, *beta, *mean, *var;
  int Cin, Cout, Kpad;
  _Float16* WT;
  float *sc, *bs;
};

static inline void launch_gemm(const _Float16* A, const _Float16* BT,
                               const float* scale, const float* bias, void* out,
                               int R, int Kpad, int Nout, int rowsPerBlock,
                               bool mp, int flags, hipStream_t s)
{
  dim3 grid(R / rowsPerBlock, Nout / 64);
  bool relu = (flags & FLAG_RELU) != 0;
  bool f32o = (flags & FLAG_F32OUT) != 0;
  if (mp) {  // pooled layers are always relu, f32 out
    if (rowsPerBlock == 32)
      wmma_gemm_t<2, true, true, true><<<grid, 256, 0, s>>>(A, BT, scale, bias, out, Kpad, Nout);
    else if (rowsPerBlock == 64)
      wmma_gemm_t<4, true, true, true><<<grid, 256, 0, s>>>(A, BT, scale, bias, out, Kpad, Nout);
    else
      wmma_gemm_t<8, true, true, true><<<grid, 256, 0, s>>>(A, BT, scale, bias, out, Kpad, Nout);
  } else if (rowsPerBlock == 32) {
    if (relu)
      wmma_gemm_t<2, false, true, false><<<grid, 256, 0, s>>>(A, BT, scale, bias, out, Kpad, Nout);
    else
      wmma_gemm_t<2, false, false, true><<<grid, 256, 0, s>>>(A, BT, scale, bias, out, Kpad, Nout);
  } else {
    if (f32o)
      wmma_gemm_t<8, false, false, true><<<grid, 256, 0, s>>>(A, BT, scale, bias, out, Kpad, Nout);
    else
      wmma_gemm_t<8, false, true, false><<<grid, 256, 0, s>>>(A, BT, scale, bias, out, Kpad, Nout);
  }
}

extern "C" void kernel_launch(void* const* d_in, const int* in_sizes, int n_in,
                              void* d_out, int out_size, void* d_ws, size_t ws_size,
                              hipStream_t stream)
{
  (void)in_sizes; (void)n_in; (void)out_size; (void)ws_size;
  const float* xyz_ctx  = (const float*)d_in[0];   // [8,2048,3]
  const float* centers  = (const float*)d_in[1];   // [8,4,3]
  const float* xyz_tgt  = (const float*)d_in[2];   // [32,1024,3]
  const float* mtok     = (const float*)d_in[111];
  const float* peW      = (const float*)d_in[112];
  const float* peB      = (const float*)d_in[113];
  const float* W1  = (const float*)d_in[114]; const float* b1  = (const float*)d_in[115];
  const float* lng = (const float*)d_in[116]; const float* lnb = (const float*)d_in[117];
  const float* W2  = (const float*)d_in[118]; const float* b2  = (const float*)d_in[119];
  const float* Wg1 = (const float*)d_in[120]; const float* bg1 = (const float*)d_in[121];
  const float* Wg2 = (const float*)d_in[122]; const float* bg2 = (const float*)d_in[123];
  float* out = (float*)d_out;

  const size_t OFF_PRED   = 0;
  const size_t OFF_TGT    = 1048576;     // 8*4*32*1024
  const size_t OFF_CTXXYZ = 2097152;
  const size_t OFF_TGTXYZ = 2097920;

  // ---- workspace bump allocator -------------------------------------------
  size_t off = 0;
  auto alloc = [&](size_t bytes) -> char* {
    char* p = (char*)d_ws + off;
    off += (bytes + 255) & ~(size_t)255;
    return p;
  };

  // ---- build 18 SA layers + BN params -------------------------------------
  LayerW Lw[18];
  for (int enc = 0; enc < 2; ++enc) {
    int pbase = enc ? 57 : 3;
    for (int L = 0; L < 3; ++L) {
      const SAConf& c = g_sa[L];
      int cins[3]  = {c.Cin, c.ch0, c.ch1};
      int couts[3] = {c.ch0, c.ch1, c.ch2};
      int kpads[3] = {c.CinPad, c.ch0, c.ch1};
      for (int j = 0; j < 3; ++j) {
        LayerW& lw = Lw[enc * 9 + L * 3 + j];
        int base = pbase + (L * 3 + j) * 6;
        lw.W    = (const float*)d_in[base + 0];
        lw.b    = (const float*)d_in[base + 1];
        lw.g    = (const float*)d_in[base + 2];
        lw.beta = (const float*)d_in[base + 3];
        lw.mean = (const float*)d_in[base + 4];
        lw.var  = (const float*)d_in[base + 5];
        lw.Cin = cins[j]; lw.Cout = couts[j]; lw.Kpad = kpads[j];
        lw.WT = (_Float16*)alloc((size_t)lw.Cout * lw.Kpad * 2);
        lw.sc = (float*)alloc((size_t)lw.Cout * 4);
        lw.bs = (float*)alloc((size_t)lw.Cout * 4);
      }
    }
  }
  // predictor transposed weights
  _Float16* W1T  = (_Float16*)alloc((size_t)2048 * 1024 * 2);
  _Float16* W2T  = (_Float16*)alloc((size_t)1024 * 2048 * 2);
  _Float16* Wg1T = (_Float16*)alloc((size_t)2048 * CONDPAD * 2);
  _Float16* Wg2T = (_Float16*)alloc((size_t)1024 * 2048 * 2);

  // big activation buffers (grouped input aliases L2 output region)
  _Float16* BUF_X = (_Float16*)alloc((size_t)134217728);
  _Float16* BUF_Y = (_Float16*)alloc((size_t)67108864);
  int*   fpsIdx = (int*)alloc((size_t)16384 * 4);
  int*   ballIdx = (int*)alloc((size_t)16384 * 32 * 4);
  float* NX[3];
  NX[0] = (float*)alloc((size_t)16384 * 3 * 4);
  NX[1] = (float*)alloc((size_t)4096 * 3 * 4);
  NX[2] = (float*)alloc((size_t)1024 * 3 * 4);
  float* Fbuf[2];
  Fbuf[0] = (float*)alloc((size_t)16384 * 128 * 4);
  Fbuf[1] = (float*)alloc((size_t)4096 * 256 * 4);
  float*    CTXF = (float*)alloc((size_t)256 * 1024 * 4);
  _Float16* ctxA = (_Float16*)alloc((size_t)256 * 1024 * 2);
  float*    H1   = (float*)alloc((size_t)256 * 2048 * 4);
  _Float16* LN16 = (_Float16*)alloc((size_t)256 * 2048 * 2);
  float*    HOUT = (float*)alloc((size_t)256 * 1024 * 4);
  _Float16* COND = (_Float16*)alloc((size_t)32 * CONDPAD * 2);
  _Float16* Z16  = (_Float16*)alloc((size_t)32 * 2048 * 2);
  float*    GPRE = (float*)alloc((size_t)32 * 1024 * 4);

  // ---- weight prep --------------------------------------------------------
  for (int i = 0; i < 18; ++i) {
    const LayerW& lw = Lw[i];
    int n = lw.Cout * lw.Kpad;
    prep_wt_kernel<<<(n + 255) / 256, 256, 0, stream>>>(lw.W, lw.Cin, lw.Cout, lw.Kpad, lw.WT);
    prep_bn_kernel<<<(lw.Cout + 255) / 256, 256, 0, stream>>>(lw.b, lw.g, lw.beta, lw.mean,
                                                              lw.var, lw.Cout, lw.sc, lw.bs);
  }
  prep_wt_kernel<<<(2048 * 1024 + 255) / 256, 256, 0, stream>>>(W1, 1024, 2048, 1024, W1T);
  prep_wt_kernel<<<(1024 * 2048 + 255) / 256, 256, 0, stream>>>(W2, 2048, 1024, 2048, W2T);
  prep_wt_kernel<<<(2048 * CONDPAD + 255) / 256, 256, 0, stream>>>(Wg1, 1048, 2048, CONDPAD, Wg1T);
  prep_wt_kernel<<<(1024 * 2048 + 255) / 256, 256, 0, stream>>>(Wg2, 2048, 1024, 2048, Wg2T);

  // ---- encoders -----------------------------------------------------------
  for (int enc = 0; enc < 2; ++enc) {
    int G  = enc ? 32 : 8;
    int N0 = enc ? 1024 : 2048;
    const float* xyz = enc ? xyz_tgt : xyz_ctx;
    int N = N0;
    const float* feats = nullptr;
    int Cf = 0;
    float* finalFeats  = enc ? (out + OFF_TGT) : CTXF;
    float* finalXyzDst = enc ? (out + OFF_TGTXYZ) : (out + OFF_CTXXYZ);

    for (int L = 0; L < 3; ++L) {
      const SAConf& c = g_sa[L];
      int GC = G * c.S;
      int R  = GC * c.K;
      fps_kernel<<<G, 256, 0, stream>>>(xyz, N, c.S, fpsIdx);
      gather_xyz_kernel<<<(GC + 255) / 256, 256, 0, stream>>>(xyz, fpsIdx, N, c.S, NX[L], GC);
      ballquery_kernel<<<(GC + 7) / 8, 256, 0, stream>>>(xyz, NX[L], N, c.S, c.K, c.r2, ballIdx, GC);
      group_kernel<<<GC * c.K, 64, 0, stream>>>(xyz, feats, NX[L], ballIdx,
                                                N, c.S, c.K, Cf, c.CinPad, BUF_X);
      const LayerW& l0 = Lw[enc * 9 + L * 3 + 0];
      const LayerW& l1 = Lw[enc * 9 + L * 3 + 1];
      const LayerW& l2 = Lw[enc * 9 + L * 3 + 2];
      launch_gemm(BUF_X, l0.WT, l0.sc, l0.bs, BUF_Y, R, c.CinPad, c.ch0, 128, false, FLAG_RELU, stream);
      launch_gemm(BUF_Y, l1.WT, l1.sc, l1.bs, BUF_X, R, c.ch0, c.ch1, 128, false, FLAG_RELU, stream);
      float* pooled = (L == 2) ? finalFeats : Fbuf[L];
      launch_gemm(BUF_X, l2.WT, l2.sc, l2.bs, pooled, R, c.ch1, c.ch2, c.K, true, FLAG_RELU, stream);
      xyz = NX[L]; N = c.S; feats = pooled; Cf = c.ch2;
    }
    int nx = G * 32 * 3;
    copy_f32_kernel<<<(nx + 255) / 256, 256, 0, stream>>>(NX[2], finalXyzDst, nx);
  }

  // ---- predictor ----------------------------------------------------------
  cvt_f16_kernel<<<(256 * 1024 + 255) / 256, 256, 0, stream>>>(CTXF, ctxA, 256 * 1024);
  launch_gemm(ctxA, W1T, nullptr, b1, H1, 256, 1024, 2048, 128, false, FLAG_F32OUT, stream);
  ln_relu_kernel<<<256, 256, 0, stream>>>(H1, lng, lnb, 2048, LN16);
  launch_gemm(LN16, W2T, nullptr, b2, HOUT, 256, 2048, 1024, 128, false, FLAG_F32OUT, stream);
  pe_kernel<<<32, 32, 0, stream>>>(centers, peW, peB, mtok, COND, 32);
  launch_gemm(COND, Wg1T, nullptr, bg1, Z16, 32, CONDPAD, 2048, 32, false, FLAG_RELU, stream);
  launch_gemm(Z16, Wg2T, nullptr, bg2, GPRE, 32, 2048, 1024, 32, false, FLAG_F32OUT, stream);
  combine_kernel<<<4096, 256, 0, stream>>>(HOUT, GPRE, out + OFF_PRED);
}
